// FashionMNIST_cINN_23742579213016
// MI455X (gfx1250) — compile-verified, matrix operand order checked by reading, not run on hardware
//
#include <hip/hip_runtime.h>
#include <cmath>
#include <cstdint>

#define NBATCH 256

typedef __attribute__((ext_vector_type(16))) _Float16 v16h;
typedef __attribute__((ext_vector_type(8)))  _Float16 v8h;
typedef __attribute__((ext_vector_type(8)))  float    v8f;

// ------------------------------------------------------------------
// numpy legacy RandomState(k).permutation(n) on device (MT19937 +
// Fisher-Yates with rk_interval rejection sampling).
// ------------------------------------------------------------------
struct MTState { unsigned m[624]; int i; };

__device__ static void mt_seed(MTState& s, unsigned seed) {
    s.m[0] = seed;
    for (int i = 1; i < 624; ++i)
        s.m[i] = 1812433253u * (s.m[i-1] ^ (s.m[i-1] >> 30)) + (unsigned)i;
    s.i = 624;
}
__device__ static unsigned mt_next(MTState& s) {
    if (s.i >= 624) {
        for (int k = 0; k < 624; ++k) {
            unsigned y = (s.m[k] & 0x80000000u) | (s.m[(k+1) % 624] & 0x7fffffffu);
            s.m[k] = s.m[(k+397) % 624] ^ (y >> 1) ^ ((y & 1u) ? 2567483615u : 0u);
        }
        s.i = 0;
    }
    unsigned y = s.m[s.i++];
    y ^= y >> 11; y ^= (y << 7) & 2636928640u; y ^= (y << 15) & 4022730752u; y ^= y >> 18;
    return y;
}
__device__ static unsigned rk_interval(MTState& s, unsigned mx) {
    unsigned mask = mx;
    mask |= mask >> 1; mask |= mask >> 2; mask |= mask >> 4; mask |= mask >> 8; mask |= mask >> 16;
    unsigned v;
    do { v = mt_next(s) & mask; } while (v > mx);
    return v;
}

__global__ void k_perms(int* __restrict__ out) {
    int t = blockIdx.x * blockDim.x + threadIdx.x;
    if (t >= 35) return;
    int seed, n, off;
    if      (t < 5)  { seed = t;      n = 2;   off = t * 2; }
    else if (t < 10) { seed = t - 5;  n = 4;   off = 10 + (t - 5)  * 4; }
    else if (t < 20) { seed = t - 10; n = 8;   off = 30 + (t - 10) * 8; }
    else             { seed = t - 20; n = 196; off = 110 + (t - 20) * 196; }
    MTState st;
    mt_seed(st, (unsigned)seed);
    int arr[196];
    for (int i = 0; i < n; ++i) arr[i] = i;
    for (int i = n - 1; i >= 1; --i) {
        int j = (int)rk_interval(st, (unsigned)i);
        int tmp = arr[i]; arr[i] = arr[j]; arr[j] = tmp;
    }
    for (int i = 0; i < n; ++i) out[off + i] = arr[i];
}

// ------------------------------------------------------------------
// Condition tensors
// ------------------------------------------------------------------
__global__ void k_condimg(const int* __restrict__ y, const float* __restrict__ sd,
                          float* __restrict__ out, int HW) {
    long long idx = (long long)blockIdx.x * blockDim.x + threadIdx.x;
    long long total = (long long)NBATCH * 2 * HW;
    if (idx >= total) return;
    int c = (int)((idx / HW) % 2);
    int n = (int)(idx / ((long long)2 * HW));
    out[idx] = c ? sd[n] : ((float)y[n] + 1.0f) * 0.1f;
}

__global__ void k_c4(const int* __restrict__ y, const float* __restrict__ sd,
                     float* __restrict__ c4) {
    int idx = blockIdx.x * blockDim.x + threadIdx.x;
    if (idx >= NBATCH * 11) return;
    int n = idx / 11, j = idx % 11;
    c4[idx] = (j < 10) ? ((y[n] == j) ? 1.0f : 0.0f) : sd[n];
}

__global__ void k_initjac(float* __restrict__ jac, float val) {
    if (threadIdx.x < NBATCH) jac[threadIdx.x] = val;
}

// ------------------------------------------------------------------
// Channel permutation: dst[n,c] = src[n, perm[c]]
// ------------------------------------------------------------------
__global__ void k_permute(const float* __restrict__ src, float* __restrict__ dst,
                          const int* __restrict__ perm, int C, int HW) {
    long long idx = (long long)blockIdx.x * blockDim.x + threadIdx.x;
    long long total = (long long)NBATCH * C * HW;
    if (idx >= total) return;
    int p = (int)(idx % HW);
    int c = (int)((idx / HW) % C);
    long long n = idx / ((long long)C * HW);
    dst[idx] = src[n * C * HW + (long long)perm[c] * HW + p];
}

// ------------------------------------------------------------------
// im2col (3x3 SAME) to f16, with optional second (condition) source,
// zero-padded to Kpad columns.
// ------------------------------------------------------------------
__global__ void k_im2col(const float* __restrict__ src0, int C0, int s0Stride,
                         const float* __restrict__ src1, int C1, int s1Stride,
                         int H, int W, int Kpad, _Float16* __restrict__ out) {
    int HW = H * W;
    long long total = (long long)NBATCH * HW * Kpad;
    long long idx = (long long)blockIdx.x * blockDim.x + threadIdx.x;
    if (idx >= total) return;
    long long m = idx / Kpad;
    int k = (int)(idx % Kpad);
    int n = (int)(m / HW);
    int pix = (int)(m % HW);
    int oh = pix / W, ow = pix % W;
    int Cin = C0 + C1;
    float val = 0.0f;
    if (k < Cin * 9) {
        int cin = k / 9, rr = k % 9;
        int h = oh + rr / 3 - 1, w = ow + rr % 3 - 1;
        if (h >= 0 && h < H && w >= 0 && w < W) {
            val = (cin < C0)
                ? src0[(long long)n * s0Stride + (long long)cin * HW + h * W + w]
                : src1[(long long)n * s1Stride + (long long)(cin - C0) * HW + h * W + w];
        }
    }
    out[idx] = (_Float16)val;
}

// f32 weight (Cout x Kreal row-major) -> f16 Bt (NcPad x Kpad), zero padded
__global__ void k_convw(const float* __restrict__ w, _Float16* __restrict__ bt,
                        int Cout, int Kreal, int NcPad, int Kpad) {
    long long total = (long long)NcPad * Kpad;
    long long idx = (long long)blockIdx.x * blockDim.x + threadIdx.x;
    if (idx >= total) return;
    int co = (int)(idx / Kpad), k = (int)(idx % Kpad);
    bt[idx] = (co < Cout && k < Kreal) ? (_Float16)w[(long long)co * Kreal + k] : (_Float16)0.0f;
}

// Build FC subnet input: [x(98) | c4(11) | zero-pad] -> f16 (NBATCH x Kpad)
__global__ void k_fcbuild(const float* __restrict__ x, int xStride,
                          const float* __restrict__ c4, _Float16* __restrict__ out,
                          int Kpad, int nx, int nc) {
    long long total = (long long)NBATCH * Kpad;
    long long idx = (long long)blockIdx.x * blockDim.x + threadIdx.x;
    if (idx >= total) return;
    int n = (int)(idx / Kpad), k = (int)(idx % Kpad);
    float v = 0.0f;
    if (k < nx)           v = x[(long long)n * xStride + k];
    else if (k < nx + nc) v = c4[n * nc + (k - nx)];
    out[idx] = (_Float16)v;
}

// ------------------------------------------------------------------
// WMMA GEMM: C[M x Nc] = A(f16, M x K) * Bt(f16, NcPad x K)^T + bias.
// The 16 x K weight panel for this blockIdx.y is staged once per
// workgroup into LDS with global_load_async_to_lds_b128 (ASYNCcnt),
// published with s_wait_asynccnt + barrier, then each of the 8 waves
// feeds its v_wmma_f32_16x16x32_f16 B-fragments from LDS (ds_load_b128)
// while streaming its own A rows from global with prefetch.
// ------------------------------------------------------------------
typedef __attribute__((address_space(3))) _Float16 lds_f16_t;

__global__ __launch_bounds__(256)
void k_gemm(const _Float16* __restrict__ A, const _Float16* __restrict__ Bt,
            const float* __restrict__ bias, float* __restrict__ outF32,
            _Float16* __restrict__ outF16,
            int M, int Nc, int K, int HW, int relu, int ld16, int useF16) {
    extern __shared__ _Float16 smem[];      // 16 x (K+8) halves, padded rows
    const int KL = K + 8;                   // LDS row stride (halves), 16B aligned
    int lane = threadIdx.x & 31;
    int wave = threadIdx.x >> 5;
    int mTile = blockIdx.x * 8 + wave;
    int nTile = blockIdx.y;

    // ---- async stage of the B panel (16 rows x K halves) into LDS ----
    {
        const _Float16* bBase = Bt + (long long)nTile * 16 * K;
        unsigned ldsBase = (unsigned)(uintptr_t)(lds_f16_t*)smem;
        int chunksPerRow = K >> 3;          // 16-byte chunks per row
        int total = chunksPerRow << 4;      // 16 rows
        for (int c = threadIdx.x; c < total; c += 256) {
            int row = c / chunksPerRow;
            int cc  = c - row * chunksPerRow;
            unsigned lds = ldsBase + (unsigned)((row * KL + cc * 8) * 2);
            const _Float16* g = bBase + (long long)row * K + cc * 8;
            asm volatile("global_load_async_to_lds_b128 %0, %1, off"
                         :: "v"(lds), "v"(g) : "memory");
        }
        asm volatile("s_wait_asynccnt 0" ::: "memory");
        __syncthreads();
    }

    if (mTile * 16 < M) {                   // uniform per wave; EXEC all-ones inside
        int m     = mTile * 16 + (lane & 15);
        int kbase = (lane >> 4) * 8;        // ISA 16-bit A/B fragment K-offset

        const _Float16* aRow = A + (long long)m * K;
        const _Float16* bRow = smem + (lane & 15) * KL;

        v8f acc = {};
        union Frag { v16h v; v8h h[2]; };
        for (int k0 = 0; k0 < K; k0 += 32) {
            if (k0 + 32 < K)                // prefetch next A chunk
                __builtin_prefetch(aRow + k0 + 32 + kbase, 0, 1);
            Frag a, b;
            a.h[0] = *(const v8h*)(aRow + k0 + kbase);
            a.h[1] = *(const v8h*)(aRow + k0 + 16 + kbase);
            b.h[0] = *(const v8h*)(bRow + k0 + kbase);        // ds_load_b128
            b.h[1] = *(const v8h*)(bRow + k0 + 16 + kbase);   // ds_load_b128
            acc = __builtin_amdgcn_wmma_f32_16x16x32_f16(
                false, a.v, false, b.v, (short)0, acc, false, false);
        }

        int col = nTile * 16 + (lane & 15);
        if (col < Nc) {
            float bv = bias ? bias[col] : 0.0f;
#pragma unroll
            for (int r = 0; r < 8; ++r) {
                int mm = mTile * 16 + r + ((lane >> 4) << 3);
                float v = acc[r] + bv;
                if (relu) v = fmaxf(v, 0.0f);
                if (useF16) {
                    outF16[(long long)mm * ld16 + col] = (_Float16)v;
                } else {
                    int samp = mm / HW;
                    int pix  = mm - samp * HW;
                    outF32[((long long)samp * Nc + col) * HW + pix] = v;
                }
            }
        }
    }
}

// ------------------------------------------------------------------
// Affine coupling: s = 0.636*atan(r[:, :l]); y = exp(s)*x + r[:, l:];
// jac[n] += sum(s).
// ------------------------------------------------------------------
__global__ void k_coupling(const float* __restrict__ x, const float* __restrict__ r,
                           float* __restrict__ y, float* __restrict__ jac,
                           int l, int HW, int xs, int rs, int ys) {
    long long idx = (long long)blockIdx.x * blockDim.x + threadIdx.x;
    long long total = (long long)NBATCH * l * HW;
    if (idx >= total) return;
    int p = (int)(idx % HW);
    int c = (int)((idx / HW) % l);
    int n = (int)(idx / ((long long)l * HW));
    float s = 0.636f * atanf(r[(long long)n * rs + (long long)c * HW + p]);
    float t = r[(long long)n * rs + (long long)(l + c) * HW + p];
    y[(long long)n * ys + (long long)c * HW + p] =
        __expf(s) * x[(long long)n * xs + (long long)c * HW + p] + t;
    atomicAdd(&jac[n], s);
}

// Haar downsample: (N,C,H,W) -> (N,4C,H/2,W/2), * 0.25
__global__ void k_haar(const float* __restrict__ src, int ss, int C, int H, int W,
                       float* __restrict__ dst, int ds) {
    int h2 = H / 2, w2 = W / 2;
    long long total = (long long)NBATCH * C * h2 * w2;
    long long idx = (long long)blockIdx.x * blockDim.x + threadIdx.x;
    if (idx >= total) return;
    int j = (int)(idx % w2);
    int i = (int)((idx / w2) % h2);
    int c = (int)((idx / ((long long)w2 * h2)) % C);
    int n = (int)(idx / ((long long)C * h2 * w2));
    const float* s0 = src + (long long)n * ss + (long long)c * H * W;
    float x00 = s0[(2*i) * W + 2*j],     x01 = s0[(2*i) * W + 2*j + 1];
    float x10 = s0[(2*i+1) * W + 2*j],   x11 = s0[(2*i+1) * W + 2*j + 1];
    float* d0 = dst + (long long)n * ds + (long long)(c * 4) * h2 * w2 + i * w2 + j;
    int st = h2 * w2;
    d0[0]      = 0.25f * (x00 + x01 + x10 + x11);
    d0[st]     = 0.25f * (x00 - x01 + x10 - x11);
    d0[2 * st] = 0.25f * (x00 + x01 - x10 - x11);
    d0[3 * st] = 0.25f * (x00 - x01 - x10 + x11);
}

// Row slice copy: dst[n, 0:cnt] = src[n, off:off+cnt]
__global__ void k_copy_rows(const float* __restrict__ src, int ss, int off, int cnt,
                            float* __restrict__ dst, int ds) {
    long long idx = (long long)blockIdx.x * blockDim.x + threadIdx.x;
    long long total = (long long)NBATCH * cnt;
    if (idx >= total) return;
    int i = (int)(idx % cnt);
    int n = (int)(idx / cnt);
    dst[(long long)n * ds + i] = src[(long long)n * ss + off + i];
}

// ==================================================================
extern "C" void kernel_launch(void* const* d_in, const int* in_sizes, int n_in,
                              void* d_out, int out_size, void* d_ws, size_t ws_size,
                              hipStream_t stream) {
    (void)in_sizes; (void)n_in; (void)out_size; (void)ws_size;

    const float* x  = (const float*)d_in[0];
    const int*   yl = (const int*)d_in[1];
    const float* sd = (const float*)d_in[2];
    // params flatten: stage start offsets; per block: p1(w1,b1,w2,b2), p2(...)
    auto prm = [&](int stage, int blk, int sub, int w) -> const float* {
        const int st[4] = {3, 43, 83, 163};
        return (const float*)d_in[st[stage] + blk * 8 + sub * 4 + w];
    };

    // ---- workspace layout ----
    char* wp = (char*)d_ws;
    auto alloc = [&](size_t bytes) -> void* {
        void* p = (void*)wp; wp += (bytes + 255) & ~(size_t)255; return p;
    };
    int*   permBuf = (int*)  alloc(3050 * sizeof(int));
    float* cond1   = (float*)alloc((size_t)NBATCH * 2 * 784 * 4);
    float* cond2   = (float*)alloc((size_t)NBATCH * 2 * 196 * 4);
    float* cond3   = (float*)alloc((size_t)NBATCH * 2 * 49  * 4);
    float* c4      = (float*)alloc((size_t)NBATCH * 11 * 4);
    float* bufA    = (float*)alloc((size_t)NBATCH * 1568 * 4);
    float* bufB    = (float*)alloc((size_t)NBATCH * 1568 * 4);
    float* bufC    = (float*)alloc((size_t)NBATCH * 1568 * 4);
    float* rbuf    = (float*)alloc((size_t)NBATCH * 1568 * 4);
    float* hid     = (float*)alloc((size_t)NBATCH * 32 * 784 * 4);
    _Float16* col16 = (_Float16*)alloc((size_t)200704 * 288 * 2); // max im2col
    _Float16* bt16  = (_Float16*)alloc((size_t)208 * 512 * 2);    // max weight panel
    _Float16* hid16 = (_Float16*)hid;                             // FC f16 intermediate

    float* zout = (float*)d_out;
    float* jac  = zout + (size_t)NBATCH * 1568;

    auto GE = [](long long t) { return dim3((unsigned)((t + 255) / 256)); };
    auto shb = [](int K) { return (size_t)16 * (K + 8) * 2; };    // LDS bytes for gemm

    // ---- constants / conditions / permutations ----
    const double lgh = std::log(16.0) + 4.0 * std::log(0.25);
    const float jacInit = (float)(196.0 * lgh + 98.0 * lgh); // haar1 + haar2 log-dets

    k_perms<<<dim3(1), dim3(64), 0, stream>>>(permBuf);
    k_condimg<<<GE((long long)NBATCH * 2 * 784), 256, 0, stream>>>(yl, sd, cond1, 784);
    k_condimg<<<GE((long long)NBATCH * 2 * 196), 256, 0, stream>>>(yl, sd, cond2, 196);
    k_condimg<<<GE((long long)NBATCH * 2 * 49),  256, 0, stream>>>(yl, sd, cond3, 49);
    k_c4<<<GE((long long)NBATCH * 11), 256, 0, stream>>>(yl, sd, c4);
    k_initjac<<<dim3(1), 256, 0, stream>>>(jac, jacInit);

    float* cur = const_cast<float*>(x);   // read-only use
    auto pick2 = [&](float* c, float*& P, float*& O) {
        float* arr[3] = {bufA, bufB, bufC};
        int k = 0;
        for (int i = 0; i < 3 && k < 2; ++i)
            if (arr[i] != c) { if (k == 0) P = arr[i]; else O = arr[i]; ++k; }
    };
    auto permPtr = [&](int stage, int blk) -> const int* {
        int off = stage == 0 ? blk * 2 : stage == 1 ? 10 + blk * 4
                 : stage == 2 ? 30 + blk * 8 : 110 + blk * 196;
        return permBuf + off;
    };

    // ---- conv subnet: conv3x3(relu) -> conv3x3 ----
    auto convSubnet = [&](const float* src0, int l, int C, const float* cond,
                          int H, int W, int hidC,
                          const float* w1, const float* b1,
                          const float* w2, const float* b2, float* rout) {
        int HW = H * W;
        int M = NBATCH * HW;
        int Kp1 = ((l + 2) * 9 + 31) & ~31;
        k_im2col<<<GE((long long)M * Kp1), 256, 0, stream>>>(
            src0, l, C * HW, cond, 2, 2 * HW, H, W, Kp1, col16);
        k_convw<<<GE((long long)hidC * Kp1), 256, 0, stream>>>(
            w1, bt16, hidC, (l + 2) * 9, hidC, Kp1);
        dim3 g1((unsigned)((M / 16 + 7) / 8), hidC / 16);
        k_gemm<<<g1, 256, shb(Kp1), stream>>>(col16, bt16, b1, hid, nullptr,
                                              M, hidC, Kp1, HW, 1, 0, 0);
        int Kp2 = hidC * 9;                 // 288 / 576 / 1152: already %32==0
        k_im2col<<<GE((long long)M * Kp2), 256, 0, stream>>>(
            hid, hidC, hidC * HW, nullptr, 0, 0, H, W, Kp2, col16);
        int NcP = (C + 15) & ~15;
        k_convw<<<GE((long long)NcP * Kp2), 256, 0, stream>>>(
            w2, bt16, C, Kp2, NcP, Kp2);
        dim3 g2((unsigned)((M / 16 + 7) / 8), NcP / 16);
        k_gemm<<<g2, 256, shb(Kp2), stream>>>(col16, bt16, b2, rout, nullptr,
                                              M, C, Kp2, HW, 0, 0, 0);
    };

    auto convGlow = [&](int stage, int blk, int C, int H, int W,
                        const float* cond, int hidC) {
        int HW = H * W, l = C / 2;
        float *P, *O; pick2(cur, P, O);
        k_permute<<<GE((long long)NBATCH * C * HW), 256, 0, stream>>>(
            cur, P, permPtr(stage, blk), C, HW);
        // subnet(concat(x2, cond), p2)
        convSubnet(P + l * HW, l, C, cond, H, W, hidC,
                   prm(stage, blk, 1, 0), prm(stage, blk, 1, 1),
                   prm(stage, blk, 1, 2), prm(stage, blk, 1, 3), rbuf);
        k_coupling<<<GE((long long)NBATCH * l * HW), 256, 0, stream>>>(
            P, rbuf, O, jac, l, HW, C * HW, C * HW, C * HW);          // -> y1
        // subnet(concat(y1, cond), p1)
        convSubnet(O, l, C, cond, H, W, hidC,
                   prm(stage, blk, 0, 0), prm(stage, blk, 0, 1),
                   prm(stage, blk, 0, 2), prm(stage, blk, 0, 3), rbuf);
        k_coupling<<<GE((long long)NBATCH * l * HW), 256, 0, stream>>>(
            P + l * HW, rbuf, O + l * HW, jac, l, HW, C * HW, C * HW, C * HW); // -> y2
        cur = O;
    };

    // ---- FC subnet: fc(relu) -> fc ----
    auto fcSubnet = [&](const float* xin,
                        const float* w1, const float* b1,
                        const float* w2, const float* b2, float* rout) {
        k_fcbuild<<<GE((long long)NBATCH * 128), 256, 0, stream>>>(
            xin, 196, c4, col16, 128, 98, 11);
        k_convw<<<GE(512LL * 128), 256, 0, stream>>>(w1, bt16, 512, 109, 512, 128);
        dim3 g1((unsigned)((NBATCH / 16 + 7) / 8), 512 / 16);
        k_gemm<<<g1, 256, shb(128), stream>>>(col16, bt16, b1, nullptr, hid16,
                                              NBATCH, 512, 128, 1, 1, 512, 1);
        k_convw<<<GE(208LL * 512), 256, 0, stream>>>(w2, bt16, 196, 512, 208, 512);
        dim3 g2((unsigned)((NBATCH / 16 + 7) / 8), 208 / 16);
        k_gemm<<<g2, 256, shb(512), stream>>>(hid16, bt16, b2, rout, nullptr,
                                              NBATCH, 196, 512, 1, 0, 0, 0);
    };

    auto fcGlow = [&](int blk) {
        float *P, *O; pick2(cur, P, O);
        k_permute<<<GE((long long)NBATCH * 196), 256, 0, stream>>>(
            cur, P, permPtr(3, blk), 196, 1);
        fcSubnet(P + 98, prm(3, blk, 1, 0), prm(3, blk, 1, 1),
                 prm(3, blk, 1, 2), prm(3, blk, 1, 3), rbuf);
        k_coupling<<<GE((long long)NBATCH * 98), 256, 0, stream>>>(
            P, rbuf, O, jac, 98, 1, 196, 196, 196);
        fcSubnet(O, prm(3, blk, 0, 0), prm(3, blk, 0, 1),
                 prm(3, blk, 0, 2), prm(3, blk, 0, 3), rbuf);
        k_coupling<<<GE((long long)NBATCH * 98), 256, 0, stream>>>(
            P + 98, rbuf, O + 98, jac, 98, 1, 196, 196, 196);
        cur = O;
    };

    // ================= forward pass =================
    // stage1: (N,2,28,28)
    for (int b = 0; b < 5; ++b) convGlow(0, b, 2, 28, 28, cond1, 32);
    // z1 = v[:,1:2]  (784 per sample) -> out cols [0,784)
    k_copy_rows<<<GE((long long)NBATCH * 784), 256, 0, stream>>>(
        cur, 1568, 784, 784, zout, 1568);
    // haar on v[:,0:1] -> (N,4,14,14)
    { float *P, *O; pick2(cur, P, O);
      k_haar<<<GE((long long)NBATCH * 1 * 196), 256, 0, stream>>>(
          cur, 1568, 1, 28, 28, P, 784);
      cur = P; }

    // stage2: (N,4,14,14)
    for (int b = 0; b < 5; ++b) convGlow(1, b, 4, 14, 14, cond2, 64);
    // z2 = v[:,2:4] (392) -> out cols [784,1176)
    k_copy_rows<<<GE((long long)NBATCH * 392), 256, 0, stream>>>(
        cur, 784, 392, 392, zout + 784, 1568);
    // haar on v[:,0:2] -> (N,8,7,7)
    { float *P, *O; pick2(cur, P, O);
      k_haar<<<GE((long long)NBATCH * 2 * 49), 256, 0, stream>>>(
          cur, 784, 2, 14, 14, P, 392);
      cur = P; }

    // stage3: (N,8,7,7)
    for (int b = 0; b < 10; ++b) convGlow(2, b, 8, 7, 7, cond3, 128);
    // z3 = v[:,4:8] (196) -> out cols [1176,1372)
    k_copy_rows<<<GE((long long)NBATCH * 196), 256, 0, stream>>>(
        cur, 392, 196, 196, zout + 1176, 1568);
    // v = v[:,0:4].reshape(n,196): compact to row-major (n,196)
    { float *P, *O; pick2(cur, P, O);
      k_copy_rows<<<GE((long long)NBATCH * 196), 256, 0, stream>>>(
          cur, 392, 0, 196, P, 196);
      cur = P; }

    // stage4: FC glow x15
    for (int b = 0; b < 15; ++b) fcGlow(b);

    // final v -> out cols [1372,1568)
    k_copy_rows<<<GE((long long)NBATCH * 196), 256, 0, stream>>>(
        cur, 196, 0, 196, zout + 1372, 1568);
}
